// SelfAttentionBlock_42442866819738
// MI455X (gfx1250) — matrix operand order repaired and checked by reading, not compiled, not run on hardware
//
#include <hip/hip_runtime.h>
#include <hip/hip_bf16.h>

// ---------------------------------------------------------------------------
// SelfAttentionBlock for MI455X (gfx1250, wave32, WMMA bf16 16x16x32)
//   B=4, C=128, H=W=64 (HW=4096), GROUPS=8, NUM_HEADS=4, head_dim=32
// ---------------------------------------------------------------------------

typedef __attribute__((ext_vector_type(16))) __bf16 v16bf;
typedef __attribute__((ext_vector_type(8)))  float  v8f;

#define HW    4096
#define CCH   128
#define NB    4
#define NH    4
#define HD    32

struct Frag32B { uint4 lo, hi; };          // 32 bytes -> v16bf
struct Pack8   { unsigned u[8]; };         // 32 bytes -> v16bf

__device__ __forceinline__ v16bf load_frag(const __hip_bfloat16* p0,
                                           const __hip_bfloat16* p1) {
    Frag32B f;
    f.lo = *(const uint4*)p0;
    f.hi = *(const uint4*)p1;
    return __builtin_bit_cast(v16bf, f);
}

__device__ __forceinline__ unsigned pack2_bf16(float a, float b) {
    unsigned ua = __float_as_uint(a), ub = __float_as_uint(b);
    ua = (ua + 0x7FFFu + ((ua >> 16) & 1u)) >> 16;   // RNE
    ub = (ub + 0x7FFFu + ((ub >> 16) & 1u)) >> 16;
    return (ua & 0xFFFFu) | (ub << 16);
}

// ---------------------------------------------------------------------------
// Kernel 0: convert fp32 weights to bf16 once
// ---------------------------------------------------------------------------
__global__ void k_wconv(const float* __restrict__ qkv_w,
                        const float* __restrict__ proj_w,
                        __hip_bfloat16* __restrict__ qkv_wb,
                        __hip_bfloat16* __restrict__ proj_wb) {
    int i = blockIdx.x * 256 + threadIdx.x;
    if (i < 3 * CCH * CCH) qkv_wb[i]  = __float2bfloat16(qkv_w[i]);
    if (i < CCH * CCH)     proj_wb[i] = __float2bfloat16(proj_w[i]);
}

// ---------------------------------------------------------------------------
// Kernel 1: GroupNorm. One block per (b, group). Writes:
//   xn   : fp32 [b][c][pos]   (residual)
//   xnT  : bf16 [b][pos][c]   (GEMM B operand)
// ---------------------------------------------------------------------------
__global__ void k_gnorm(const float* __restrict__ x,
                        const float* __restrict__ gw,
                        const float* __restrict__ gb,
                        float* __restrict__ xn,
                        __hip_bfloat16* __restrict__ xnT) {
    const int bg = blockIdx.x, b = bg >> 3, g = bg & 7;
    const int N = 16 * HW;
    const size_t base = ((size_t)b * CCH + g * 16) * HW;

    float s = 0.f, s2 = 0.f;
    for (int i = threadIdx.x; i < N; i += 256) {
        float v = x[base + i];
        s += v; s2 += v * v;
    }
    __shared__ float sh[512];
    sh[threadIdx.x] = s; sh[256 + threadIdx.x] = s2;
    __syncthreads();
    for (int off = 128; off; off >>= 1) {
        if ((int)threadIdx.x < off) {
            sh[threadIdx.x]       += sh[threadIdx.x + off];
            sh[256 + threadIdx.x] += sh[256 + threadIdx.x + off];
        }
        __syncthreads();
    }
    const float mean = sh[0] / (float)N;
    const float var  = sh[256] / (float)N - mean * mean;
    const float rstd = rsqrtf(var + 1e-5f);

    for (int i = threadIdx.x; i < N; i += 256) {
        int cl = i >> 12, pos = i & (HW - 1), c = g * 16 + cl;
        float v = (x[base + i] - mean) * rstd * gw[c] + gb[c];
        xn[base + i] = v;
        xnT[((size_t)b * HW + pos) * CCH + c] = __float2bfloat16(v);
    }
}

// ---------------------------------------------------------------------------
// Kernel 2: QKV GEMM, qkv[b,o,pos] = sum_c W[o,c] * xn[b,c,pos] + bias[o]
// One wave per 16(o) x 16(pos) tile, K=128 as 4 x (K=32) bf16 WMMAs.
// Outputs (bf16): qT/kT : [b][h][pos][c_local]   v : [b][h][c_local][pos]
// ---------------------------------------------------------------------------
__global__ void k_qkv(const __hip_bfloat16* __restrict__ Wb,
                      const float* __restrict__ bias,
                      const __hip_bfloat16* __restrict__ xnT,
                      __hip_bfloat16* __restrict__ qT,
                      __hip_bfloat16* __restrict__ kT,
                      __hip_bfloat16* __restrict__ vbuf) {
    const int lane = threadIdx.x & 31;
    const int warp = threadIdx.x >> 5;
    const int job  = blockIdx.x * 4 + warp;              // NB*24*256 jobs
    const int b    = job / (24 * 256);
    const int rem  = job - b * 24 * 256;
    const int ot   = rem >> 8;                           // o-tile 0..23
    const int nt   = rem & 255;                          // pos-tile 0..255

    const int obase = ot * 16, nbase = nt * 16;
    const int l16   = lane & 15;
    const int chalf = (lane < 16) ? 0 : 8;               // A row K-half / D row half
    const int arow  = obase + l16;

    const __hip_bfloat16* xb = xnT + (size_t)b * HW * CCH;
    const int pos = nbase + l16;
    const int bcol = (lane < 16) ? 0 : 8;                // B K-half

    v8f acc = {};
#pragma unroll
    for (int kk = 0; kk < 4; ++kk) {
        const int k0 = kk * 32;
        v16bf a = load_frag(Wb + (size_t)arow * CCH + k0 + chalf,
                            Wb + (size_t)arow * CCH + k0 + chalf + 16);
        v16bf bb = load_frag(xb + (size_t)pos * CCH + k0 + bcol,
                             xb + (size_t)pos * CCH + k0 + bcol + 16);
        acc = __builtin_amdgcn_wmma_f32_16x16x32_bf16(false, a, false, bb,
                                                      (short)0, acc, false, false);
    }
#pragma unroll
    for (int j = 0; j < 8; ++j)
        acc[j] += bias[obase + j + chalf];

    const int third = obase >> 7;                        // 0=q 1=k 2=v
    const int orow  = obase & 127;
    const int h     = orow >> 5;
    const int cb    = (orow & 31) + chalf;
    const size_t bh = (size_t)b * NH + h;

    if (third < 2) {
        __hip_bfloat16* dst = (third == 0 ? qT : kT) + (bh * HW + pos) * HD + cb;
        uint4 pk;
        pk.x = pack2_bf16(acc[0], acc[1]);
        pk.y = pack2_bf16(acc[2], acc[3]);
        pk.z = pack2_bf16(acc[4], acc[5]);
        pk.w = pack2_bf16(acc[6], acc[7]);
        *(uint4*)dst = pk;
    } else {
#pragma unroll
        for (int j = 0; j < 8; ++j)
            vbuf[(bh * HD + cb + j) * HW + pos] = __float2bfloat16(acc[j]);
    }
}

// ---------------------------------------------------------------------------
// Kernel 3: flash attention. Block = 4 waves = 64 queries of one (b,h).
// Per wave: Sᵀ = Kᵀ(16e x 32c) x Q(32c x 16d)  [2 WMMAs / 32 keys]
//           O += V(16c x 32e) x Pᵀ(32e x 16d)  [2 WMMAs / 32 keys]
// Online softmax state per lane (lane%16 == query column d).
// Output: aoutT bf16 [b][pos][c_global] for the proj GEMM.
// ---------------------------------------------------------------------------
__global__ void k_attn(const __hip_bfloat16* __restrict__ qT,
                       const __hip_bfloat16* __restrict__ kT,
                       const __hip_bfloat16* __restrict__ vbuf,
                       __hip_bfloat16* __restrict__ aoutT) {
    const int lane = threadIdx.x & 31;
    const int warp = threadIdx.x >> 5;
    const int wg   = blockIdx.x;                 // NB*NH*(HW/64)
    const int qb   = wg & 63;
    const int bh   = wg >> 6;
    const int b    = bh >> 2, h = bh & 3;
    const int D    = qb * 64 + warp * 16;        // query base of this wave

    const __hip_bfloat16* qbh = qT   + (size_t)bh * HW * HD;
    const __hip_bfloat16* kbh = kT   + (size_t)bh * HW * HD;
    const __hip_bfloat16* vbh = vbuf + (size_t)bh * HD * HW;

    const int   l16   = lane & 15;
    const bool  sel   = (lane < 16);
    const float scale = 0.088388347648318447f;   // 128^-0.5

    // Q fragment (B operand), loaded once: lane -> column d = D+l16
    const int qc = sel ? 0 : 8;
    const v16bf qfrag = load_frag(qbh + (size_t)(D + l16) * HD + qc,
                                  qbh + (size_t)(D + l16) * HD + qc + 16);

    v8f o0 = {}, o1 = {};
    float m = -3.0e38f, l = 0.f;

    const int aoff = sel ? 0 : 8;                // A-fragment K-half offset

    for (int E = 0; E < HW; E += 32) {
        // ---- Sᵀ tiles: e = E..E+15 (t0), E+16..E+31 (t1) -----------------
        const int erow = E + l16;
        v16bf ka0 = load_frag(kbh + (size_t)erow * HD + aoff,
                              kbh + (size_t)erow * HD + aoff + 16);
        v16bf ka1 = load_frag(kbh + (size_t)(erow + 16) * HD + aoff,
                              kbh + (size_t)(erow + 16) * HD + aoff + 16);
        v8f s0 = {}, s1 = {};
        s0 = __builtin_amdgcn_wmma_f32_16x16x32_bf16(false, ka0, false, qfrag,
                                                     (short)0, s0, false, false);
        s1 = __builtin_amdgcn_wmma_f32_16x16x32_bf16(false, ka1, false, qfrag,
                                                     (short)0, s1, false, false);

        // ---- online softmax over the 32 new keys -------------------------
        float tmax = -3.0e38f;
#pragma unroll
        for (int j = 0; j < 8; ++j) {
            s0[j] *= scale; s1[j] *= scale;
            tmax = fmaxf(tmax, fmaxf(s0[j], s1[j]));
        }
        tmax = fmaxf(tmax, __shfl_xor(tmax, 16, 32));
        const float m_new = fmaxf(m, tmax);
        const float corr  = __expf(m - m_new);
        float rs = 0.f;
#pragma unroll
        for (int j = 0; j < 8; ++j) {
            s0[j] = __expf(s0[j] - m_new);
            s1[j] = __expf(s1[j] - m_new);
            rs += s0[j] + s1[j];
        }
        rs += __shfl_xor(rs, 16, 32);
        l = l * corr + rs;
        m = m_new;
#pragma unroll
        for (int j = 0; j < 8; ++j) { o0[j] *= corr; o1[j] *= corr; }

        // ---- Pᵀ (C layout) -> B layout via xor-16 shuffle + bf16 pack ----
        Pack8 pk;
#pragma unroll
        for (int v2 = 0; v2 < 8; ++v2) {
            const int j0 = (v2 < 4) ? 2 * v2 : 2 * v2 - 8;
            const int j1 = j0 + 1;
            float send0 = sel ? s1[j0] : s0[j0];
            float send1 = sel ? s1[j1] : s0[j1];
            float own0  = sel ? s0[j0] : s1[j0];
            float own1  = sel ? s0[j1] : s1[j1];
            float r0 = __shfl_xor(send0, 16, 32);
            float r1 = __shfl_xor(send1, 16, 32);
            const bool use_own = ((v2 < 4) == sel);
            pk.u[v2] = pack2_bf16(use_own ? own0 : r0, use_own ? own1 : r1);
        }
        const v16bf pfrag = __builtin_bit_cast(v16bf, pk);

        // ---- O += V x Pᵀ --------------------------------------------------
        v16bf va0 = load_frag(vbh + (size_t)l16 * HW + E + (sel ? 0 : 16),
                              vbh + (size_t)l16 * HW + E + (sel ? 0 : 16) + 8);
        v16bf va1 = load_frag(vbh + (size_t)(l16 + 16) * HW + E + (sel ? 0 : 16),
                              vbh + (size_t)(l16 + 16) * HW + E + (sel ? 0 : 16) + 8);
        o0 = __builtin_amdgcn_wmma_f32_16x16x32_bf16(false, va0, false, pfrag,
                                                     (short)0, o0, false, false);
        o1 = __builtin_amdgcn_wmma_f32_16x16x32_bf16(false, va1, false, pfrag,
                                                     (short)0, o1, false, false);
    }

    // ---- normalize and store bf16 to aoutT[b][pos][h*32 + c_local] -------
    const float inv_l = 1.0f / l;
    const int d = D + l16;
    const int ch = sel ? 0 : 8;
    __hip_bfloat16* dst = aoutT + ((size_t)b * HW + d) * CCH + h * HD;
#pragma unroll
    for (int j = 0; j < 8; ++j) {
        dst[ch + j]      = __float2bfloat16(o0[j] * inv_l);
        dst[16 + ch + j] = __float2bfloat16(o1[j] * inv_l);
    }
}

// ---------------------------------------------------------------------------
// Kernel 4: proj GEMM + bias + residual:
//   out[b,o,pos] = sum_c Wp[o,c]*aout[b,c,pos] + pb[o] + xn[b,o,pos]
// ---------------------------------------------------------------------------
__global__ void k_proj(const __hip_bfloat16* __restrict__ Wpb,
                       const float* __restrict__ pb,
                       const __hip_bfloat16* __restrict__ aoutT,
                       const float* __restrict__ xn,
                       float* __restrict__ out) {
    const int lane = threadIdx.x & 31;
    const int warp = threadIdx.x >> 5;
    const int job  = blockIdx.x * 4 + warp;       // NB*8*256 jobs
    const int b    = job / (8 * 256);
    const int rem  = job - b * 8 * 256;
    const int ot   = rem >> 8;
    const int nt   = rem & 255;

    const int obase = ot * 16, nbase = nt * 16;
    const int l16   = lane & 15;
    const int chalf = (lane < 16) ? 0 : 8;
    const int arow  = obase + l16;
    const int pos   = nbase + l16;
    const int bcol  = (lane < 16) ? 0 : 8;

    const __hip_bfloat16* ab = aoutT + (size_t)b * HW * CCH;

    v8f acc = {};
#pragma unroll
    for (int kk = 0; kk < 4; ++kk) {
        const int k0 = kk * 32;
        v16bf a = load_frag(Wpb + (size_t)arow * CCH + k0 + chalf,
                            Wpb + (size_t)arow * CCH + k0 + chalf + 16);
        v16bf bb = load_frag(ab + (size_t)pos * CCH + k0 + bcol,
                             ab + (size_t)pos * CCH + k0 + bcol + 16);
        acc = __builtin_amdgcn_wmma_f32_16x16x32_bf16(false, a, false, bb,
                                                      (short)0, acc, false, false);
    }
#pragma unroll
    for (int j = 0; j < 8; ++j) {
        const int o = obase + j + chalf;
        const size_t idx = ((size_t)b * CCH + o) * HW + pos;
        out[idx] = acc[j] + pb[o] + xn[idx];
    }
}

// ---------------------------------------------------------------------------
// Launch
// ---------------------------------------------------------------------------
extern "C" void kernel_launch(void* const* d_in, const int* in_sizes, int n_in,
                              void* d_out, int out_size, void* d_ws, size_t ws_size,
                              hipStream_t stream) {
    const float* x      = (const float*)d_in[0];
    const float* gn_w   = (const float*)d_in[1];
    const float* gn_b   = (const float*)d_in[2];
    const float* qkv_w  = (const float*)d_in[3];
    const float* qkv_b  = (const float*)d_in[4];
    const float* proj_w = (const float*)d_in[5];
    const float* proj_b = (const float*)d_in[6];
    float* out = (float*)d_out;

    char* ws = (char*)d_ws;
    size_t off = 0;
    float*          xn      = (float*)(ws + off);          off += (size_t)NB*CCH*HW*4;   // 8 MB
    __hip_bfloat16* xnT     = (__hip_bfloat16*)(ws + off); off += (size_t)NB*HW*CCH*2;   // 4 MB
    __hip_bfloat16* qT      = (__hip_bfloat16*)(ws + off); off += (size_t)NB*NH*HW*HD*2; // 4 MB
    __hip_bfloat16* kT      = (__hip_bfloat16*)(ws + off); off += (size_t)NB*NH*HW*HD*2; // 4 MB
    __hip_bfloat16* vbuf    = (__hip_bfloat16*)(ws + off); off += (size_t)NB*NH*HD*HW*2; // 4 MB
    __hip_bfloat16* aoutT   = (__hip_bfloat16*)(ws + off); off += (size_t)NB*HW*CCH*2;   // 4 MB
    __hip_bfloat16* qkv_wb  = (__hip_bfloat16*)(ws + off); off += (size_t)3*CCH*CCH*2;
    __hip_bfloat16* proj_wb = (__hip_bfloat16*)(ws + off); off += (size_t)CCH*CCH*2;

    k_wconv<<<192, 256, 0, stream>>>(qkv_w, proj_w, qkv_wb, proj_wb);
    k_gnorm<<<NB * 8, 256, 0, stream>>>(x, gn_w, gn_b, xn, xnT);
    k_qkv  <<<NB * 24 * 256 / 4, 128, 0, stream>>>(qkv_wb, qkv_b, xnT, qT, kT, vbuf);
    k_attn <<<NB * NH * (HW / 64), 128, 0, stream>>>(qT, kT, vbuf, aoutT);
    k_proj <<<NB * 8 * 256 / 4, 128, 0, stream>>>(proj_wb, proj_b, aoutT, xn, out);
}
